// MultiHeadAttentionBlock_84018150244882
// MI455X (gfx1250) — matrix operand order repaired and checked by reading, not compiled, not run on hardware
//
#include <hip/hip_runtime.h>
#include <hip/hip_bf16.h>
#include <cstdint>

#define DMODEL 1024
#define NHEAD  16
#define DKH    64
#define NBATCH 2
#define SEQLEN 2048

typedef __attribute__((ext_vector_type(16))) __bf16 v16bf;
typedef __attribute__((ext_vector_type(8)))  __bf16 v8bf;
typedef __attribute__((ext_vector_type(4)))  __bf16 v4bf;
typedef __attribute__((ext_vector_type(8)))  float  v8f;
typedef int b128_t __attribute__((vector_size(16)));   // matches builtin param type

// Async Global->LDS (ASYNCcnt) path: probe-gated, device pass only.
#if defined(__gfx1250__) && defined(__HIP_DEVICE_COMPILE__) && \
    __has_builtin(__builtin_amdgcn_global_load_async_to_lds_b128) && \
    __has_builtin(__builtin_amdgcn_s_wait_asynccnt)
#define HAVE_ASYNC_LDS 1
#else
#define HAVE_ASYNC_LDS 0
#endif

__device__ __forceinline__ v8f wmma_bf16(const v16bf& a, const v16bf& b, const v8f& c) {
  // 8 args: (neg_a, A, neg_b, B, c_mod, C, reuse_a, reuse_b)
  return __builtin_amdgcn_wmma_f32_16x16x32_bf16(false, a, false, b, (short)0, c, false, false);
}

__device__ __forceinline__ v16bf cat8(v8bf lo, v8bf hi) {
  return __builtin_shufflevector(lo, hi, 0, 1, 2, 3, 4, 5, 6, 7, 8, 9, 10, 11, 12, 13, 14, 15);
}

__device__ __forceinline__ v8f vzero8() {
  v8f z;
#pragma unroll
  for (int i = 0; i < 8; ++i) z[i] = 0.0f;
  return z;
}

// Stage one 16B (8 x bf16) chunk into LDS.
// fp32 source: load 2x float4, convert, ds_store_b128.
// bf16 source: async DMA straight into LDS when available (no VGPR bounce).
__device__ __forceinline__ void stage8(const float* g, __bf16* l) {
  const float4 f0 = *reinterpret_cast<const float4*>(g);
  const float4 f1 = *reinterpret_cast<const float4*>(g + 4);
  v8bf h;
  h[0] = (__bf16)f0.x; h[1] = (__bf16)f0.y; h[2] = (__bf16)f0.z; h[3] = (__bf16)f0.w;
  h[4] = (__bf16)f1.x; h[5] = (__bf16)f1.y; h[6] = (__bf16)f1.z; h[7] = (__bf16)f1.w;
  *reinterpret_cast<v8bf*>(l) = h;
}
__device__ __forceinline__ void stage8(const __bf16* g, __bf16* l) {
#if HAVE_ASYNC_LDS
  b128_t* gt = (b128_t*)(void*)g;   // pointee-type change in generic AS
  b128_t* lt = (b128_t*)(void*)l;
  __builtin_amdgcn_global_load_async_to_lds_b128(
      (__attribute__((address_space(1))) b128_t*)gt,   // AS-only cast
      (__attribute__((address_space(3))) b128_t*)lt, 0, 0);
#else
  *reinterpret_cast<v8bf*>(l) = *reinterpret_cast<const v8bf*>(g);
#endif
}
__device__ __forceinline__ void stage_wait() {
#if HAVE_ASYNC_LDS
  __builtin_amdgcn_s_wait_asynccnt(0);
#endif
}

__device__ __forceinline__ void store_out(float* p, float v)  { *p = v; }
__device__ __forceinline__ void store_out(__bf16* p, float v) { *p = (__bf16)v; }

// A-fragment (16x32 bf16, ISA 7.12.2): two contiguous 8-elem chunks per lane.
//   elems 0..7  -> K = base + half*8 + 0..7
//   elems 8..15 -> K = base + 16 + half*8 + 0..7
__device__ __forceinline__ v16bf load_afrag(const __bf16* row, int base, int half) {
  v8bf lo = *reinterpret_cast<const v8bf*>(row + base + half * 8);
  v8bf hi = *reinterpret_cast<const v8bf*>(row + base + 16 + half * 8);
  return cat8(lo, hi);
}
// B-fragment (32x16 bf16) from a transposed [n][k] LDS tile: same K order as A.
//   elems 0..7  -> K = base + half*8 + 0..7
//   elems 8..15 -> K = base + 16 + half*8 + 0..7
__device__ __forceinline__ v16bf load_bfrag(const __bf16* nrow, int base, int half) {
  v8bf lo = *reinterpret_cast<const v8bf*>(nrow + base + half * 8);
  v8bf hi = *reinterpret_cast<const v8bf*>(nrow + base + 16 + half * 8);
  return cat8(lo, hi);
}

// ---------------------------------------------------------------------------
// GEMM: C[M,N] = A[M,K] * Bw[K,N].  Tile 128(M) x 64(N) x 64(K), 256 thr =
// 8 waves; wave w -> 16x64 strip.  A row-major bf16 in LDS (async-copied when
// the source is already bf16); B staged TRANSPOSED ([n][k]) so B fragments
// are contiguous ds_load_b128 pairs.
// ---------------------------------------------------------------------------
template <typename AT, typename OT>
__global__ __launch_bounds__(256) void gemm_bf16_wmma(
    const AT* __restrict__ A, const float* __restrict__ Bw,
    OT* __restrict__ C, int M, int N, int K)
{
  __shared__ __bf16 As[128][72];   // 64 + 8 pad (144B rows, 16B aligned)
  __shared__ __bf16 Bst[64][72];   // transposed: [n][k]

  const int t    = threadIdx.x;
  const int lane = t & 31;
  const int wave = t >> 5;
  const int half = (lane >> 4) & 1;
  const int l16  = lane & 15;
  const int row0 = blockIdx.y * 128;
  const int n0   = blockIdx.x * 64;

  v8f acc[4];
#pragma unroll
  for (int i = 0; i < 4; ++i) acc[i] = vzero8();

  for (int kb = 0; kb < K; kb += 64) {
    // Stage A tile (128x64): 1024 8-elem chunks / 256 threads.
#pragma unroll
    for (int p = 0; p < 4; ++p) {
      const int lin = t + p * 256;
      const int r = lin >> 3, c8 = lin & 7;
      stage8(&A[(size_t)(row0 + r) * K + kb + c8 * 8], &As[r][c8 * 8]);
    }
    // Stage B tile (64k x 64n) TRANSPOSED: lane-coalesced fp32 reads along n,
    // per-thread contiguous along k -> v4bf stores into Bst[n][k].
    {
      const int n = t & 63, tg = t >> 6;   // tg in 0..3
#pragma unroll
      for (int p = 0; p < 4; ++p) {
        const int k0 = (tg + p * 4) * 4;   // covers k = 0..63
        v4bf hv;
#pragma unroll
        for (int j = 0; j < 4; ++j)
          hv[j] = (__bf16)Bw[(size_t)(kb + k0 + j) * N + n0 + n];
        *reinterpret_cast<v4bf*>(&Bst[n][k0]) = hv;
      }
    }
    if (kb + 64 < K)  // warm next A tile (global_prefetch_b8)
      __builtin_prefetch(&A[(size_t)(row0 + (t >> 1)) * K + kb + 64], 0, 1);
    stage_wait();
    __syncthreads();

    const int ml = wave * 16 + l16;
#pragma unroll
    for (int c = 0; c < 2; ++c) {
      const v16bf a = load_afrag(&As[ml][0], c * 32, half);
#pragma unroll
      for (int nt = 0; nt < 4; ++nt) {
        const v16bf b = load_bfrag(&Bst[nt * 16 + l16][0], c * 32, half);
        acc[nt] = wmma_bf16(a, b, acc[nt]);
      }
    }
    __syncthreads();
  }

#pragma unroll
  for (int nt = 0; nt < 4; ++nt)
#pragma unroll
    for (int r = 0; r < 8; ++r) {
      const int m = row0 + wave * 16 + half * 8 + r;
      const int n = n0 + nt * 16 + l16;
      store_out(&C[(size_t)m * N + n], acc[nt][r]);
    }
}

// ---------------------------------------------------------------------------
// Flash attention over one (batch, head, 64-query tile).  128 thr = 4 waves.
// Q/K tiles are straight bf16 copies -> async Global->LDS when available;
// V is staged transposed [d][key] via d-coalesced reads; P bounces through
// wave-private LDS guarded by s_wait_dscnt (no block barrier needed).
// ---------------------------------------------------------------------------
__global__ __launch_bounds__(128) void attn_flash(
    const __bf16* __restrict__ Qp,   // [B*S][DMODEL] bf16
    const __bf16* __restrict__ Kp,
    const __bf16* __restrict__ Vp,
    const int*    __restrict__ msk,  // [B][S][S]
    __bf16*       __restrict__ Xo)   // [B*S][DMODEL] bf16 (heads merged)
{
  __shared__ __bf16 Qs[64][72];      // [q][d]      A-layout source
  __shared__ __bf16 Ks[64][72];      // [key][d]    B-layout source for scores
  __shared__ __bf16 Vt[64][72];      // [d][key]    B-layout source for P.V
  __shared__ __bf16 Ps[4][16][72];   // wave-private P, A-layout source

  const int t    = threadIdx.x;
  const int lane = t & 31;
  const int wave = t >> 5;
  const int half = (lane >> 4) & 1;
  const int l16  = lane & 15;

  const int q0 = blockIdx.x * 64;
  const int h  = blockIdx.y;
  const int b  = blockIdx.z;
  const size_t rowbase = (size_t)b * SEQLEN;
  const int hc = h * DKH;

  // Stage Q tile once: 512 8-elem chunks / 128 threads.
#pragma unroll
  for (int p = 0; p < 4; ++p) {
    const int lin = t + p * 128;
    const int r = lin >> 3, c8 = lin & 7;
    stage8(&Qp[(rowbase + q0 + r) * DMODEL + hc + c8 * 8], &Qs[r][c8 * 8]);
  }
  stage_wait();
  __syncthreads();

  const int ml = wave * 16 + l16;
  v16bf qa[2];
#pragma unroll
  for (int c = 0; c < 2; ++c) qa[c] = load_afrag(&Qs[ml][0], c * 32, half);

  float mrow[8], lrow[8];
  v8f o[4];
#pragma unroll
  for (int r = 0; r < 8; ++r) { mrow[r] = -INFINITY; lrow[r] = 0.0f; }
#pragma unroll
  for (int i = 0; i < 4; ++i) o[i] = vzero8();

  const int qrow_base = q0 + wave * 16 + half * 8;

  for (int kt = 0; kt < SEQLEN / 64; ++kt) {
    const int k0 = kt * 64;
    __syncthreads();  // previous iteration's Ks/Vt reads are done

    // K tile: straight row copy (keys x d) -- async DMA when available.
#pragma unroll
    for (int p = 0; p < 4; ++p) {
      const int lin = t + p * 128;
      const int key = lin >> 3, c8 = lin & 7;
      stage8(&Kp[(rowbase + k0 + key) * DMODEL + hc + c8 * 8], &Ks[key][c8 * 8]);
    }
    // V tile transposed: lanes cover consecutive d (coalesced), each thread
    // gathers 4 keys at 2KB stride and stores contiguous along key.
    {
      const int d = t & 63, tg = t >> 6;  // tg in 0..1
#pragma unroll
      for (int p = 0; p < 8; ++p) {
        const int key0 = (tg + p * 2) * 4;
        v4bf hv;
#pragma unroll
        for (int j = 0; j < 4; ++j)
          hv[j] = Vp[(rowbase + k0 + key0 + j) * DMODEL + hc + d];
        *reinterpret_cast<v4bf*>(&Vt[d][key0]) = hv;
      }
    }
    stage_wait();
    __syncthreads();

    // scores = Q . K^T  (16q x 64key per wave)
    v8f s[4];
#pragma unroll
    for (int nt = 0; nt < 4; ++nt) {
      s[nt] = vzero8();
#pragma unroll
      for (int c = 0; c < 2; ++c) {
        const v16bf bf = load_bfrag(&Ks[nt * 16 + l16][0], c * 32, half);
        s[nt] = wmma_bf16(qa[c], bf, s[nt]);
      }
    }

    // scale (1/sqrt(64)) + mask
#pragma unroll
    for (int nt = 0; nt < 4; ++nt) {
      const int kcol = k0 + nt * 16 + l16;
#pragma unroll
      for (int r = 0; r < 8; ++r) {
        const float sv = s[nt][r] * 0.125f;
        const int mv = msk[((size_t)b * SEQLEN + (qrow_base + r)) * SEQLEN + kcol];
        s[nt][r] = (mv == 0) ? -1.0e9f : sv;
      }
    }

    // online softmax: each row lives in one 16-lane half at vector elem r
#pragma unroll
    for (int r = 0; r < 8; ++r) {
      float mx = fmaxf(fmaxf(s[0][r], s[1][r]), fmaxf(s[2][r], s[3][r]));
#pragma unroll
      for (int d = 1; d < 16; d <<= 1) mx = fmaxf(mx, __shfl_xor(mx, d, 32));
      const float mnew = fmaxf(mrow[r], mx);
      const float corr = __expf(mrow[r] - mnew);
      float rs = 0.0f;
#pragma unroll
      for (int nt = 0; nt < 4; ++nt) {
        const float pv = __expf(s[nt][r] - mnew);
        s[nt][r] = pv;
        rs += pv;
      }
#pragma unroll
      for (int d = 1; d < 16; d <<= 1) rs += __shfl_xor(rs, d, 32);
      lrow[r] = lrow[r] * corr + rs;
      mrow[r] = mnew;
#pragma unroll
      for (int nt = 0; nt < 4; ++nt) o[nt][r] *= corr;
    }

    // P: C-layout regs -> wave-private LDS (bf16) -> A-layout b128 reads.
#pragma unroll
    for (int nt = 0; nt < 4; ++nt)
#pragma unroll
      for (int r = 0; r < 8; ++r)
        Ps[wave][half * 8 + r][nt * 16 + l16] = (__bf16)s[nt][r];
    asm volatile("s_wait_dscnt 0" ::: "memory");  // wave-private: no barrier

    // O += P . V
#pragma unroll
    for (int c = 0; c < 2; ++c) {
      const v16bf pa = load_afrag(&Ps[wave][l16][0], c * 32, half);
#pragma unroll
      for (int nt = 0; nt < 4; ++nt) {
        const v16bf bf = load_bfrag(&Vt[nt * 16 + l16][0], c * 32, half);
        o[nt] = wmma_bf16(pa, bf, o[nt]);
      }
    }
  }

  // epilogue: divide by row sum, merge heads via column offset
#pragma unroll
  for (int nt = 0; nt < 4; ++nt)
#pragma unroll
    for (int r = 0; r < 8; ++r) {
      const int q = qrow_base + r;
      Xo[(rowbase + q) * DMODEL + hc + nt * 16 + l16] = (__bf16)(o[nt][r] / lrow[r]);
    }
}

// ---------------------------------------------------------------------------
extern "C" void kernel_launch(void* const* d_in, const int* in_sizes, int n_in,
                              void* d_out, int out_size, void* d_ws, size_t ws_size,
                              hipStream_t stream) {
  const float* q   = (const float*)d_in[0];
  const float* k   = (const float*)d_in[1];
  const float* v   = (const float*)d_in[2];
  const int*   msk = (const int*)d_in[3];
  const float* wq  = (const float*)d_in[4];
  const float* wk  = (const float*)d_in[5];
  const float* wv  = (const float*)d_in[6];
  const float* wo  = (const float*)d_in[7];

  const size_t MM = (size_t)NBATCH * SEQLEN;  // 4096 rows
  __bf16* Qp = (__bf16*)d_ws;                 // 8 MB each
  __bf16* Kp = Qp + MM * DMODEL;
  __bf16* Vp = Kp + MM * DMODEL;
  __bf16* Xo = Vp + MM * DMODEL;              // 32 MB total workspace

  dim3 gg(DMODEL / 64, (unsigned)(MM / 128));
  gemm_bf16_wmma<float, __bf16><<<gg, 256, 0, stream>>>(q, wq, Qp, (int)MM, DMODEL, DMODEL);
  gemm_bf16_wmma<float, __bf16><<<gg, 256, 0, stream>>>(k, wk, Kp, (int)MM, DMODEL, DMODEL);
  gemm_bf16_wmma<float, __bf16><<<gg, 256, 0, stream>>>(v, wv, Vp, (int)MM, DMODEL, DMODEL);

  attn_flash<<<dim3(SEQLEN / 64, NHEAD, NBATCH), 128, 0, stream>>>(Qp, Kp, Vp, msk, Xo);

  gemm_bf16_wmma<__bf16, float><<<gg, 256, 0, stream>>>(Xo, wo, (float*)d_out, (int)MM, DMODEL, DMODEL);
}
